// MultiHeadAttention_58875411694296
// MI455X (gfx1250) — compile-verified, matrix operand order checked by reading, not run on hardware
//
#include <hip/hip_runtime.h>
#include <stdint.h>

typedef __bf16 bf16_t;
typedef __attribute__((ext_vector_type(16))) __bf16 v16bf;
typedef __attribute__((ext_vector_type(8))) float v8f;

#define T_SEQ  512
#define NHEADS 16
#define KCH    64
#define CDIM   1024
#define BATCH  8
#define MROWS  (BATCH * T_SEQ) /* 4096 */
#define RLW    64              /* padded rel-logit row width: delta -31..31 -> j=delta+31 */

__device__ __forceinline__ bf16_t f2bf(float f) {
  uint32_t u = __builtin_bit_cast(uint32_t, f);
  u += 0x7FFFu + ((u >> 16) & 1u);            // round-to-nearest-even
  uint16_t h = (uint16_t)(u >> 16);
  return __builtin_bit_cast(bf16_t, h);
}
__device__ __forceinline__ float bf2f(bf16_t b) {
  uint16_t h = __builtin_bit_cast(uint16_t, b);
  uint32_t u = ((uint32_t)h) << 16;
  return __builtin_bit_cast(float, u);
}

union Frag {
  v16bf v;
  uint32_t u[8];
};

// 16x32 bf16 A-fragment (also B-fragment when loading from B^T, row-major, ld even).
// ISA 7.12.2: lanes 0-15 row m=lane, K pairs {0,1..6,7} in v0-3, {16..23} in v4-7;
// lanes 16-31 same rows, K ranges +8.  Pairs are contiguous -> compiler merges to b128.
__device__ __forceinline__ v16bf load_frag(const bf16_t* p, int ld) {
  int lane = threadIdx.x & 31;
  int m = lane & 15, hi = lane >> 4;
  Frag f;
#pragma unroll
  for (int r = 0; r < 8; ++r) {
    int k = 2 * r + (r >= 4 ? 8 : 0) + 8 * hi;
    f.u[r] = *reinterpret_cast<const uint32_t*>(p + m * ld + k);
  }
  return f.v;
}

__device__ __forceinline__ v8f wmma_bf16(v16bf a, v16bf b, v8f c) {
  return __builtin_amdgcn_wmma_f32_16x16x32_bf16(false, a, false, b, (short)0, c,
                                                 false, false);
}

// ---------------- conversion kernels ----------------
__global__ void cvt_bf16_kernel(const float* __restrict__ in, bf16_t* __restrict__ out,
                                int n) {
  int i = blockIdx.x * blockDim.x + threadIdx.x;
  if (i < n) out[i] = f2bf(in[i]);
}

__global__ void transpose_w_kernel(const float* __restrict__ W, bf16_t* __restrict__ Wt) {
  int i = blockIdx.x * blockDim.x + threadIdx.x;  // CDIM*CDIM threads
  int k = i >> 10, n = i & 1023;
  Wt[(size_t)n * CDIM + k] = f2bf(W[(size_t)k * CDIM + n]);
}

// ---------------- GEMM: out = A[4096x1024] @ W + bias ----------------
// Per wave: 32 rows x 128 cols, two accumulator banks share each B fragment.
// MODE 0: Q (scaled 0.125) -> bf16 [bh][t][d]
// MODE 1: K               -> bf16 [bh][t][d]
// MODE 2: V               -> bf16 [bh][d][t]  (transposed)
// MODE 3: final           -> fp32 [row][n] (d_out)
template <int MODE>
__global__ __launch_bounds__(256, 4) void gemm_kernel(const bf16_t* __restrict__ A,
                                                      const bf16_t* __restrict__ Wt,
                                                      const float* __restrict__ bias,
                                                      bf16_t* __restrict__ obf,
                                                      float* __restrict__ of32) {
  const int K = CDIM;
  int lane = threadIdx.x & 31;
  int w = threadIdx.x >> 5;
  int rbase = blockIdx.x * 256 + w * 32;
  int nbase = blockIdx.y * 128;

  v8f acc[2][8] = {};
  const bf16_t* arow0 = A + (size_t)rbase * K;
  const bf16_t* arow1 = arow0 + (size_t)16 * K;
  for (int kk = 0; kk < K; kk += 32) {
    v16bf a0 = load_frag(arow0 + kk, K);
    v16bf a1 = load_frag(arow1 + kk, K);
#pragma unroll
    for (int nt = 0; nt < 8; ++nt) {
      v16bf b = load_frag(Wt + (size_t)(nbase + nt * 16) * K + kk, K);
      acc[0][nt] = wmma_bf16(a0, b, acc[0][nt]);
      acc[1][nt] = wmma_bf16(a1, b, acc[1][nt]);
    }
  }

  int hi = lane >> 4, col = lane & 15;
#pragma unroll
  for (int g = 0; g < 2; ++g) {
#pragma unroll
    for (int nt = 0; nt < 8; ++nt) {
      int n0 = nbase + nt * 16 + col;
      float bv = bias[n0];
#pragma unroll
      for (int v = 0; v < 8; ++v) {
        int row = rbase + g * 16 + v + 8 * hi;
        float val = acc[g][nt][v] + bv;
        if (MODE == 3) {
          of32[(size_t)row * CDIM + n0] = val;
        } else {
          int b = row >> 9, t = row & 511, h = n0 >> 6, d = n0 & 63;
          if (MODE == 0) {
            obf[((size_t)(b * NHEADS + h) * T_SEQ + t) * KCH + d] = f2bf(val * 0.125f);
          } else if (MODE == 1) {
            obf[((size_t)(b * NHEADS + h) * T_SEQ + t) * KCH + d] = f2bf(val);
          } else {
            obf[((size_t)(b * NHEADS + h) * KCH + d) * T_SEQ + t] = f2bf(val);
          }
        }
      }
    }
  }
}

// ------- padded rel-k logit table: rl2[bh*512+t][j] = Qs[bh][t][:].erk[j-31+4][:] ----
// j = delta+31, nonzero only for |delta|<=4; zeros elsewhere so the band add in the
// attention kernel is branchless within the three diagonal s-tiles.
__global__ void relk_kernel(const bf16_t* __restrict__ Qs, const float* __restrict__ erk,
                            float* __restrict__ rl2) {
  int idx = blockIdx.x * blockDim.x + threadIdx.x;
  if (idx >= BATCH * NHEADS * T_SEQ * RLW) return;
  int j = idx & (RLW - 1);
  int row = idx >> 6;  // bh*512 + t
  int delta = j - 31;
  float s = 0.f;
  if (delta >= -4 && delta <= 4) {
    const bf16_t* q = Qs + (size_t)row * KCH;
    const float* e = erk + (delta + 4) * KCH;
#pragma unroll
    for (int d = 0; d < KCH; ++d) s += bf2f(q[d]) * e[d];
  }
  rl2[idx] = s;  // Qs already carries inv_sqrt
}

// ---------------- fused attention (two-chunk online softmax, <256 VGPRs) -------------
// 128 KB LDS/WG -> 2 WGs per WGP (320 KB); 16 waves/WGP = 4 waves/SIMD, matching the
// waves-per-eu hint that caps allocation at 256 VGPRs (no s_set_vgpr_msb traffic).
__global__ __launch_bounds__(256, 4) void attn_kernel(const bf16_t* __restrict__ Qs,
                                                      const bf16_t* __restrict__ Kb,
                                                      const bf16_t* __restrict__ Vt,
                                                      const float* __restrict__ rl2,
                                                      const float* __restrict__ erv,
                                                      bf16_t* __restrict__ Obf) {
  extern __shared__ bf16_t lds[];  // 8 waves * 16 rows * 512 bf16 = 128 KB (Pexp)
  int lane = threadIdx.x & 31, w = threadIdx.x >> 5;
  int bh = blockIdx.y;
  int t0 = blockIdx.x * 128 + w * 16;
  int t0u = __builtin_amdgcn_readfirstlane(t0);  // wave-uniform -> scalar branches
  int st0 = t0u >> 4;
  int hi = lane >> 4, col = lane & 15;
  bf16_t* P = lds + (size_t)w * 16 * T_SEQ;

  const bf16_t* qptr = Qs + ((size_t)bh * T_SEQ + t0) * KCH;
  v16bf aq0 = load_frag(qptr, KCH);
  v16bf aq1 = load_frag(qptr + 32, KCH);

  const float* rlrow = rl2 + (size_t)bh * T_SEQ * RLW;
  const bf16_t* kbase = Kb + (size_t)bh * T_SEQ * KCH;
  const bf16_t* vbase = Vt + (size_t)bh * KCH * T_SEQ;

  float m_run[8], sum_loc[8] = {}, corr[8];
  v8f acc_o[4] = {};
#pragma unroll
  for (int v = 0; v < 8; ++v) { m_run[v] = -3.0e38f; corr[v] = 0.f; }

#pragma unroll
  for (int half = 0; half < 2; ++half) {
    // ---- scores for 16 s-tiles of this half, fp32, in registers (128 VGPRs) ----
    v8f sc[16];
#pragma unroll
    for (int i = 0; i < 16; ++i) {
      int st = half * 16 + i;
      const bf16_t* kptr = kbase + (size_t)st * 16 * KCH;
      v8f acc = {};
      acc = wmma_bf16(aq0, load_frag(kptr, KCH), acc);
      acc = wmma_bf16(aq1, load_frag(kptr + 32, KCH), acc);
      // banded rel-k logits: only the three tiles straddling the diagonal
      if (st >= st0 - 1 && st <= st0 + 1) {
        int s = st * 16 + col;
#pragma unroll
        for (int v = 0; v < 8; ++v) {
          int t = t0 + v + 8 * hi;
          acc[v] += rlrow[t * RLW + (s - t) + 31];  // zero outside |s-t|<=4
        }
      }
      sc[i] = acc;
    }

    // ---- merge max, rescale running sum + partial O ----
    float lm[8];
#pragma unroll
    for (int v = 0; v < 8; ++v) lm[v] = sc[0][v];
#pragma unroll
    for (int i = 1; i < 16; ++i)
#pragma unroll
      for (int v = 0; v < 8; ++v) lm[v] = fmaxf(lm[v], sc[i][v]);
#pragma unroll
    for (int v = 0; v < 8; ++v)
#pragma unroll
      for (int off = 1; off < 16; off <<= 1)
        lm[v] = fmaxf(lm[v], __shfl_xor(lm[v], off, 32));
#pragma unroll
    for (int v = 0; v < 8; ++v) {
      float mnew = fmaxf(m_run[v], lm[v]);
      float c = __expf(m_run[v] - mnew);  // 0 on first half, rescale factor on second
      m_run[v] = mnew;
      sum_loc[v] *= c;
      corr[v] = c;  // after half==1 this is the stale-P correction for cols < 256
#pragma unroll
      for (int nd = 0; nd < 4; ++nd) acc_o[nd][v] *= c;
    }

    // ---- exp in registers, single bf16 store of Pexp to LDS ----
#pragma unroll
    for (int i = 0; i < 16; ++i) {
      int s = (half * 16 + i) * 16 + col;
#pragma unroll
      for (int v = 0; v < 8; ++v) {
        float e = __expf(sc[i][v] - m_run[v]);
        sum_loc[v] += e;
        P[(v + 8 * hi) * T_SEQ + s] = f2bf(e);
      }
    }

    // ---- partial O += Pexp(half) @ V(half rows); A-fragments preloaded ----
    v16bf af[8];
#pragma unroll
    for (int ks = 0; ks < 8; ++ks)
      af[ks] = load_frag(P + (half * 8 + ks) * 32, T_SEQ);
#pragma unroll
    for (int nd = 0; nd < 4; ++nd) {
      const bf16_t* vptr = vbase + (size_t)nd * 16 * T_SEQ + half * 256;
#pragma unroll
      for (int ks = 0; ks < 8; ++ks)
        acc_o[nd] = wmma_bf16(af[ks], load_frag(vptr + ks * 32, T_SEQ), acc_o[nd]);
    }
  }

  // ---- finalize: sum reduce, banded rel-v (stale half-0 P corrected), 1/sum ----
#pragma unroll
  for (int v = 0; v < 8; ++v)
#pragma unroll
    for (int off = 1; off < 16; off <<= 1)
      sum_loc[v] += __shfl_xor(sum_loc[v], off, 32);
  float rinv[8];
#pragma unroll
  for (int v = 0; v < 8; ++v) rinv[v] = 1.0f / sum_loc[v];

  int b = bh >> 4, h = bh & 15;
#pragma unroll
  for (int nd = 0; nd < 4; ++nd) {
    int d = nd * 16 + col;
    float ev[9];
#pragma unroll
    for (int j = 0; j < 9; ++j) ev[j] = erv[j * KCH + d];
#pragma unroll
    for (int v = 0; v < 8; ++v) {
      int r = v + 8 * hi;
      int t = t0 + r;
      float o = acc_o[nd][v];
#pragma unroll
      for (int j = 0; j < 9; ++j) {
        int sp = t + j - 4;
        if (sp >= 0 && sp < T_SEQ) {
          float pv = bf2f(P[r * T_SEQ + sp]);
          o += pv * (sp < 256 ? corr[v] : 1.0f) * ev[j];
        }
      }
      o *= rinv[v];
      Obf[((size_t)b * T_SEQ + t) * CDIM + h * KCH + d] = f2bf(o);
    }
  }
}

// ---------------- host launch ----------------
extern "C" void kernel_launch(void* const* d_in, const int* in_sizes, int n_in,
                              void* d_out, int out_size, void* d_ws, size_t ws_size,
                              hipStream_t stream) {
  (void)in_sizes; (void)n_in; (void)out_size; (void)ws_size;
  const float* x   = (const float*)d_in[0];
  const float* c   = (const float*)d_in[1];
  const float* wq  = (const float*)d_in[2];
  const float* bq  = (const float*)d_in[3];
  const float* wk  = (const float*)d_in[4];
  const float* bk  = (const float*)d_in[5];
  const float* wv  = (const float*)d_in[6];
  const float* bv  = (const float*)d_in[7];
  const float* wo  = (const float*)d_in[8];
  const float* bo  = (const float*)d_in[9];
  const float* erk = (const float*)d_in[10];
  const float* erv = (const float*)d_in[11];
  float* out = (float*)d_out;

  size_t off = 0;
  auto alloc = [&](size_t bytes) -> char* {
    char* p = (char*)d_ws + off;
    off += (bytes + 255) & ~(size_t)255;
    return p;
  };
  const size_t NTOK = (size_t)MROWS * CDIM;  // 4M elements
  bf16_t* xbf = (bf16_t*)alloc(NTOK * 2);
  bf16_t* cbf = (bf16_t*)alloc(NTOK * 2);
  bf16_t* wqt = (bf16_t*)alloc((size_t)CDIM * CDIM * 2);
  bf16_t* wkt = (bf16_t*)alloc((size_t)CDIM * CDIM * 2);
  bf16_t* wvt = (bf16_t*)alloc((size_t)CDIM * CDIM * 2);
  bf16_t* wot = (bf16_t*)alloc((size_t)CDIM * CDIM * 2);
  bf16_t* Qsb = (bf16_t*)alloc(NTOK * 2);
  bf16_t* Kbb = (bf16_t*)alloc(NTOK * 2);
  bf16_t* Vtb = (bf16_t*)alloc(NTOK * 2);
  float*  rlg = (float*)alloc((size_t)BATCH * NHEADS * T_SEQ * RLW * 4);
  bf16_t* Obf = (bf16_t*)alloc(NTOK * 2);

  cvt_bf16_kernel<<<(int)((NTOK + 255) / 256), 256, 0, stream>>>(x, xbf, (int)NTOK);
  cvt_bf16_kernel<<<(int)((NTOK + 255) / 256), 256, 0, stream>>>(c, cbf, (int)NTOK);
  int wgrid = (CDIM * CDIM) / 256;
  transpose_w_kernel<<<wgrid, 256, 0, stream>>>(wq, wqt);
  transpose_w_kernel<<<wgrid, 256, 0, stream>>>(wk, wkt);
  transpose_w_kernel<<<wgrid, 256, 0, stream>>>(wv, wvt);
  transpose_w_kernel<<<wgrid, 256, 0, stream>>>(wo, wot);

  dim3 gg(MROWS / 256, CDIM / 128);
  gemm_kernel<0><<<gg, 256, 0, stream>>>(xbf, wqt, bq, Qsb, nullptr);
  gemm_kernel<1><<<gg, 256, 0, stream>>>(cbf, wkt, bk, Kbb, nullptr);
  gemm_kernel<2><<<gg, 256, 0, stream>>>(cbf, wvt, bv, Vtb, nullptr);

  int nrl = BATCH * NHEADS * T_SEQ * RLW;
  relk_kernel<<<(nrl + 255) / 256, 256, 0, stream>>>(Qsb, erk, rlg);

  dim3 ga(T_SEQ / 128, BATCH * NHEADS);
  attn_kernel<<<ga, 256, 8 * 16 * T_SEQ * (int)sizeof(bf16_t), stream>>>(
      Qsb, Kbb, Vtb, rlg, erv, Obf);

  gemm_kernel<3><<<gg, 256, 0, stream>>>(Obf, wot, bo, nullptr, out);
}
